// QSelfAttention_77549929496778
// MI455X (gfx1250) — matrix operand order repaired and checked by reading, not compiled, not run on hardware
//
#include <hip/hip_runtime.h>
#include <math.h>
#include <stdint.h>

// ---------------------------------------------------------------------------
// Types for CDNA5 WMMA (wave32): bf16 16x16x32, f32 accumulate
// ---------------------------------------------------------------------------
typedef __attribute__((ext_vector_type(16))) __bf16 bf16x16;
typedef __attribute__((ext_vector_type(8)))  float  f32x8;

struct FragU { uint32_t u[8]; };

static __device__ __forceinline__ bf16x16 frag_cast(const FragU& f) {
  return __builtin_bit_cast(bf16x16, f);
}

static __device__ __forceinline__ uint16_t f32_to_bf16(float f) {
  union { float f; uint32_t u; } x; x.f = f;
  uint32_t r = x.u + 0x7FFFu + ((x.u >> 16) & 1u);  // round-nearest-even
  return (uint16_t)(r >> 16);
}

static __device__ __forceinline__ f32x8 wmma_bf16(const FragU& a, const FragU& b, f32x8 c) {
  return __builtin_amdgcn_wmma_f32_16x16x32_bf16(
      /*neg_a=*/false, frag_cast(a), /*neg_b=*/false, frag_cast(b),
      /*c_mod=*/(short)0, c, /*reuse_a=*/false, /*reuse_b=*/false);
}

#define B_   2
#define L_   2048
#define C_   1024
#define H_   16
#define D_   64
#define MAX_SCALE_MUL 4.605170185988092f
#define ATT_SCALE 0.25f

// ---------------------------------------------------------------------------
// Kernel: f32 -> bf16 elementwise (n multiple of 4)
// ---------------------------------------------------------------------------
__global__ void cvt_bf16_kernel(const float* __restrict__ in,
                                uint16_t* __restrict__ out, int n) {
  int i = (blockIdx.x * blockDim.x + threadIdx.x) * 4;
  if (i < n) {
    float4 v = *(const float4*)(in + i);
    out[i + 0] = f32_to_bf16(v.x);
    out[i + 1] = f32_to_bf16(v.y);
    out[i + 2] = f32_to_bf16(v.z);
    out[i + 3] = f32_to_bf16(v.w);
  }
}

// Fused qkv bias vector: [q_bias, zeros, v_bias]
__global__ void build_bias_kernel(const float* __restrict__ qb,
                                  const float* __restrict__ vb,
                                  float* __restrict__ bias) {
  int i = blockIdx.x * blockDim.x + threadIdx.x;
  if (i < 3 * C_) {
    float v = 0.0f;
    if (i < C_)            v = qb[i];
    else if (i >= 2 * C_)  v = vb[i - 2 * C_];
    bias[i] = v;
  }
}

// ---------------------------------------------------------------------------
// Kernel: C[M,N] (f32) = A[M,K](bf16) @ W[N,K]^T(bf16) + bias[N]
// Block = 256 threads = 8 waves (4Mx2N); block tile 256x128;
// wave tile 64x64 = 4x4 fragments (128 acc VGPRs).
// per K-step(32): 4 A-frags + 4 B-frags (16 b128 loads) -> 16 WMMA  (1.0 ld/wmma)
// ---------------------------------------------------------------------------
__global__ __launch_bounds__(256) void gemm_bf16_kernel(
    const uint16_t* __restrict__ A, const uint16_t* __restrict__ W,
    const float* __restrict__ bias, float* __restrict__ C,
    int M, int N, int K) {
  const int tid  = threadIdx.x;
  const int wave = tid >> 5;
  const int lane = tid & 31;
  const int half = lane >> 4;      // 0: lanes 0-15, 1: lanes 16-31
  const int ln   = lane & 15;

  const int wy = wave >> 1;        // 0..3
  const int wx = wave & 1;         // 0..1
  const int rowBase = blockIdx.y * 256 + wy * 64;
  const int colBase = blockIdx.x * 128 + wx * 64;

  f32x8 acc[4][4];
#pragma unroll
  for (int mt = 0; mt < 4; ++mt)
#pragma unroll
    for (int nt = 0; nt < 4; ++nt)
#pragma unroll
      for (int e = 0; e < 8; ++e) acc[mt][nt][e] = 0.0f;

  const int akb = half ? 16 : 0;   // A-frag K offset per half-wave
  const int bkb = half ? 16 : 0;   // B-frag K offset per half-wave

  for (int k0 = 0; k0 < K; k0 += 32) {
    FragU a[4], b[4];
#pragma unroll
    for (int mt = 0; mt < 4; ++mt) {
      const uint16_t* ap = A + (size_t)(rowBase + mt * 16 + ln) * K + k0;
#pragma unroll
      for (int j = 0; j < 4; ++j) {
        a[mt].u[j]     = *(const uint32_t*)(ap + akb + 2 * j);
        a[mt].u[4 + j] = *(const uint32_t*)(ap + 8 + akb + 2 * j);
      }
    }
#pragma unroll
    for (int nt = 0; nt < 4; ++nt) {
      const uint16_t* wp = W + (size_t)(colBase + nt * 16 + ln) * K + k0 + bkb;
#pragma unroll
      for (int j = 0; j < 8; ++j)
        b[nt].u[j] = *(const uint32_t*)(wp + 2 * j);
    }
#pragma unroll
    for (int mt = 0; mt < 4; ++mt)
#pragma unroll
      for (int nt = 0; nt < 4; ++nt)
        acc[mt][nt] = wmma_bf16(a[mt], b[nt], acc[mt][nt]);
  }

#pragma unroll
  for (int nt = 0; nt < 4; ++nt) {
    const int col = colBase + nt * 16 + ln;
    const float bv = bias ? bias[col] : 0.0f;
#pragma unroll
    for (int mt = 0; mt < 4; ++mt) {
#pragma unroll
      for (int r = 0; r < 8; ++r) {
        const int row = rowBase + mt * 16 + half * 8 + r;
        C[(size_t)row * N + col] = acc[mt][nt][r] + bv;
      }
    }
  }
}

// ---------------------------------------------------------------------------
// Kernel: l2norm + scale + pack to head-major bf16 q/k and transposed v
// qkv_raw: [B*L, 3*C]; qh/kh: [B,H,L,D]; vth: [B,H,D,L]
// One wave per (b,h,l) row; lane handles d=lane and d=lane+32.
// ---------------------------------------------------------------------------
__global__ __launch_bounds__(256) void norm_pack_kernel(
    const float* __restrict__ qkv, const float* __restrict__ scale_mul,
    uint16_t* __restrict__ qh, uint16_t* __restrict__ kh,
    uint16_t* __restrict__ vth) {
  const int wave = threadIdx.x >> 5;
  const int lane = threadIdx.x & 31;
  const int idx  = blockIdx.x * 8 + wave;   // over B*H*L
  const int bh = idx >> 11;
  const int l  = idx & (L_ - 1);
  const int b  = bh >> 4;
  const int h  = bh & 15;

  const size_t rowq = ((size_t)(b * L_ + l)) * (3 * C_) + (size_t)h * D_;
  const float q0 = qkv[rowq + lane],           q1 = qkv[rowq + 32 + lane];
  const float k0 = qkv[rowq + C_ + lane],      k1 = qkv[rowq + C_ + 32 + lane];
  const float v0 = qkv[rowq + 2 * C_ + lane],  v1 = qkv[rowq + 2 * C_ + 32 + lane];

  float sq = q0 * q0 + q1 * q1;
  float sk = k0 * k0 + k1 * k1;
#pragma unroll
  for (int m = 16; m >= 1; m >>= 1) {
    sq += __shfl_xor(sq, m, 32);
    sk += __shfl_xor(sk, m, 32);
  }
  const float invq = 1.0f / fmaxf(sqrtf(sq), 1e-12f);
  const float invk = 1.0f / fmaxf(sqrtf(sk), 1e-12f);
  // fold softmax-scale and per-head logit scale into q
  const float sm = __expf(fminf(scale_mul[h], MAX_SCALE_MUL)) * invq * ATT_SCALE;

  const size_t obase = ((size_t)bh * L_ + l) * D_;
  qh[obase + lane]      = f32_to_bf16(q0 * sm);
  qh[obase + 32 + lane] = f32_to_bf16(q1 * sm);
  kh[obase + lane]      = f32_to_bf16(k0 * invk);
  kh[obase + 32 + lane] = f32_to_bf16(k1 * invk);
  vth[((size_t)bh * D_ + lane)      * L_ + l] = f32_to_bf16(v0);
  vth[((size_t)bh * D_ + 32 + lane) * L_ + l] = f32_to_bf16(v1);
}

// ---------------------------------------------------------------------------
// Kernel: flash attention. Block = 256 thr = 8 waves; each wave owns 16 q rows.
// Streams keys in chunks of 32: S (2 tiles x 2 WMMA) -> +bias -> online softmax
// (half-wave shfl_xor row reductions) -> P via LDS relayout -> O += P@V (4 WMMA).
// ---------------------------------------------------------------------------
__global__ __launch_bounds__(256) void attn_kernel(
    const uint16_t* __restrict__ qh, const uint16_t* __restrict__ kh,
    const uint16_t* __restrict__ vth, const float* __restrict__ bias2d,
    uint16_t* __restrict__ oh) {
  __shared__ uint16_t lds_p[8 * 16 * 32];   // 1KB per wave

  const int tid  = threadIdx.x;
  const int wave = tid >> 5;
  const int lane = tid & 31;
  const int half = lane >> 4;
  const int ln   = lane & 15;

  const int bh = blockIdx.y;
  const int b  = bh >> 4;
  const int h  = bh & 15;
  const int rowBase = blockIdx.x * 128 + wave * 16;   // q row within [0,L)

  const uint16_t* qp = qh  + (size_t)bh * L_ * D_;
  const uint16_t* kp = kh  + (size_t)bh * L_ * D_;
  const uint16_t* vp = vth + (size_t)bh * D_ * L_;

  // Q A-fragments (16 rows x D=64 -> 2 frags of 16x32), loaded once
  FragU qa[2];
  {
    const int akb = half ? 16 : 0;
    const uint16_t* qrow = qp + (size_t)(rowBase + ln) * D_;
#pragma unroll
    for (int j = 0; j < 4; ++j) {
      qa[0].u[j]     = *(const uint32_t*)(qrow + akb + 2 * j);
      qa[0].u[4 + j] = *(const uint32_t*)(qrow + 8 + akb + 2 * j);
      qa[1].u[j]     = *(const uint32_t*)(qrow + 32 + akb + 2 * j);
      qa[1].u[4 + j] = *(const uint32_t*)(qrow + 40 + akb + 2 * j);
    }
  }

  f32x8 oacc[4];
#pragma unroll
  for (int dt = 0; dt < 4; ++dt)
#pragma unroll
    for (int e = 0; e < 8; ++e) oacc[dt][e] = 0.0f;

  float run_max[8], run_sum[8];
#pragma unroll
  for (int r = 0; r < 8; ++r) { run_max[r] = -INFINITY; run_sum[r] = 0.0f; }

  const int dkb = half ? 16 : 0;   // B-frag K(d) offset per half-wave
  uint16_t* myp = lds_p + wave * 512;

  for (int kc = 0; kc < L_; kc += 32) {
    // --- S = Q @ K^T : two 16x16 tiles (keys kc..+15 and kc+16..+31) ---
    FragU kb0a, kb0b, kb1a, kb1b;
    {
      const uint16_t* kr0 = kp + (size_t)(kc + ln) * D_ + dkb;
      const uint16_t* kr1 = kp + (size_t)(kc + 16 + ln) * D_ + dkb;
#pragma unroll
      for (int j = 0; j < 8; ++j) {
        kb0a.u[j] = *(const uint32_t*)(kr0 + 2 * j);        // d in [0,32)
        kb0b.u[j] = *(const uint32_t*)(kr0 + 32 + 2 * j);   // d in [32,64)
        kb1a.u[j] = *(const uint32_t*)(kr1 + 2 * j);
        kb1b.u[j] = *(const uint32_t*)(kr1 + 32 + 2 * j);
      }
    }
    f32x8 zc;
#pragma unroll
    for (int e = 0; e < 8; ++e) zc[e] = 0.0f;
    f32x8 s0 = wmma_bf16(qa[0], kb0a, zc);
    s0 = wmma_bf16(qa[1], kb0b, s0);
    f32x8 s1 = wmma_bf16(qa[0], kb1a, zc);
    s1 = wmma_bf16(qa[1], kb1b, s1);

    // --- add attention bias (L2-resident, re-read per head) ---
    const int qr0 = rowBase + half * 8;
#pragma unroll
    for (int r = 0; r < 8; ++r) {
      const float* bp = bias2d + (size_t)(qr0 + r) * L_ + kc + ln;
      s0[r] += bp[0];
      s1[r] += bp[16];
    }

    // --- online softmax: per-row max/sum via half-wave xor reductions ---
    float alpha[8];
#pragma unroll
    for (int r = 0; r < 8; ++r) {
      float mv = fmaxf(s0[r], s1[r]);
#pragma unroll
      for (int m = 8; m >= 1; m >>= 1) mv = fmaxf(mv, __shfl_xor(mv, m, 32));
      const float nm = fmaxf(run_max[r], mv);
      alpha[r] = __expf(run_max[r] - nm);
      const float p0 = __expf(s0[r] - nm);
      const float p1 = __expf(s1[r] - nm);
      float rs = p0 + p1;
#pragma unroll
      for (int m = 8; m >= 1; m >>= 1) rs += __shfl_xor(rs, m, 32);
      run_sum[r] = run_sum[r] * alpha[r] + rs;
      run_max[r] = nm;
      // stash P (bf16) in this wave's LDS slice, row-major 16x32
      myp[(half * 8 + r) * 32 + ln]      = f32_to_bf16(p0);
      myp[(half * 8 + r) * 32 + 16 + ln] = f32_to_bf16(p1);
    }
#pragma unroll
    for (int dt = 0; dt < 4; ++dt)
#pragma unroll
      for (int r = 0; r < 8; ++r) oacc[dt][r] *= alpha[r];

    asm volatile("s_wait_dscnt 0" ::: "memory");

    // --- re-layout P as 16x32 A-fragment from LDS ---
    FragU pa;
    {
      const int akb = half ? 16 : 0;
      const uint16_t* prow = myp + ln * 32;
#pragma unroll
      for (int j = 0; j < 4; ++j) {
        pa.u[j]     = *(const uint32_t*)(prow + akb + 2 * j);
        pa.u[4 + j] = *(const uint32_t*)(prow + 8 + akb + 2 * j);
      }
    }

    // --- O += P @ V : V^T stored [D,L] row-major -> standard B-frag loads ---
#pragma unroll
    for (int dt = 0; dt < 4; ++dt) {
      FragU vb;
      const uint16_t* vr = vp + (size_t)(dt * 16 + ln) * L_ + kc + dkb;
#pragma unroll
      for (int j = 0; j < 8; ++j)
        vb.u[j] = *(const uint32_t*)(vr + 2 * j);
      oacc[dt] = wmma_bf16(pa, vb, oacc[dt]);
    }
  }

  // --- finalize: O /= rowsum; write [B,L,H,D] bf16 for the proj GEMM ---
  float inv[8];
#pragma unroll
  for (int r = 0; r < 8; ++r) inv[r] = 1.0f / run_sum[r];
#pragma unroll
  for (int dt = 0; dt < 4; ++dt) {
#pragma unroll
    for (int r = 0; r < 8; ++r) {
      const int gr  = b * L_ + rowBase + half * 8 + r;
      const int col = h * D_ + dt * 16 + ln;
      oh[(size_t)gr * C_ + col] = f32_to_bf16(oacc[dt][r] * inv[r]);
    }
  }
}

// ---------------------------------------------------------------------------
// Host-side orchestration
// ---------------------------------------------------------------------------
extern "C" void kernel_launch(void* const* d_in, const int* in_sizes, int n_in,
                              void* d_out, int out_size, void* d_ws, size_t ws_size,
                              hipStream_t stream) {
  (void)in_sizes; (void)n_in; (void)out_size; (void)ws_size;
  const float* x         = (const float*)d_in[0];
  const float* attn_bias = (const float*)d_in[1];
  const float* w_qkv     = (const float*)d_in[2];
  const float* q_bias    = (const float*)d_in[3];
  const float* v_bias    = (const float*)d_in[4];
  const float* scale_mul = (const float*)d_in[5];
  const float* w_proj    = (const float*)d_in[6];
  const float* b_proj    = (const float*)d_in[7];
  float* out = (float*)d_out;

  char* w = (char*)d_ws;
  size_t off = 0;
  auto take = [&](size_t bytes) -> char* {
    char* p = w + off;
    off = (off + bytes + 255) & ~(size_t)255;
    return p;
  };
  const size_t ML = (size_t)B_ * L_;                 // 4096
  uint16_t* xh      = (uint16_t*)take(ML * C_ * 2);            // x bf16
  uint16_t* wqkvh   = (uint16_t*)take((size_t)3 * C_ * C_ * 2);
  uint16_t* wprojh  = (uint16_t*)take((size_t)C_ * C_ * 2);
  float*    biasq   = (float*)   take((size_t)3 * C_ * 4);
  float*    qkv_raw = (float*)   take(ML * 3 * C_ * 4);        // 48 MiB
  uint16_t* qh      = (uint16_t*)take((size_t)B_ * H_ * L_ * D_ * 2);
  uint16_t* kh      = (uint16_t*)take((size_t)B_ * H_ * L_ * D_ * 2);
  uint16_t* vth     = (uint16_t*)take((size_t)B_ * H_ * D_ * L_ * 2);
  uint16_t* oh      = (uint16_t*)take(ML * C_ * 2);

  // 1) down-convert activations & weights to bf16
  {
    int n = (int)(ML * C_);
    cvt_bf16_kernel<<<(n / 4 + 255) / 256, 256, 0, stream>>>(x, xh, n);
    n = 3 * C_ * C_;
    cvt_bf16_kernel<<<(n / 4 + 255) / 256, 256, 0, stream>>>(w_qkv, wqkvh, n);
    n = C_ * C_;
    cvt_bf16_kernel<<<(n / 4 + 255) / 256, 256, 0, stream>>>(w_proj, wprojh, n);
  }
  build_bias_kernel<<<(3 * C_ + 255) / 256, 256, 0, stream>>>(q_bias, v_bias, biasq);

  // 2) QKV GEMM: [4096,3072] = xh @ wqkvh^T + biasq
  {
    dim3 grid((3 * C_) / 128, (int)(ML / 256));
    gemm_bf16_kernel<<<grid, 256, 0, stream>>>(xh, wqkvh, biasq, qkv_raw,
                                               (int)ML, 3 * C_, C_);
  }

  // 3) l2norm + scale + pack (q/k head-major, v transposed)
  norm_pack_kernel<<<(B_ * H_ * L_) / 8, 256, 0, stream>>>(qkv_raw, scale_mul,
                                                           qh, kh, vth);

  // 4) flash attention
  {
    dim3 grid(L_ / 128, B_ * H_);
    attn_kernel<<<grid, 256, 0, stream>>>(qh, kh, vth, attn_bias, oh);
  }

  // 5) output projection: out = oh @ wprojh^T + b_proj (f32 out)
  {
    dim3 grid(C_ / 128, (int)(ML / 256));
    gemm_bf16_kernel<<<grid, 256, 0, stream>>>(oh, wprojh, b_proj, out,
                                               (int)ML, C_, C_);
  }
}
